// FprojNN_21921513079375
// MI455X (gfx1250) — compile-verified, matrix operand order checked by reading, not run on hardware
//
#include <hip/hip_runtime.h>
#include <hip/hip_bf16.h>
#include <math.h>

// ---------------------------------------------------------------------------
// FprojNN for MI455X (gfx1250): bf16 WMMA MLP, SVD in VALU, LDS ping-pong.
// GELU via CDNA5 hardware v_tanh_f32. Both n-tiles fused per wave so each
// A-fragment is read from LDS once and feeds two WMMAs (halves LDS traffic).
// ---------------------------------------------------------------------------

typedef __bf16 bf16_t;
typedef __attribute__((ext_vector_type(16))) __bf16 bfrag;   // A/B fragment (16x32 / 32x16 bf16)
typedef __attribute__((ext_vector_type(8)))  __bf16 bhalf8;  // 16-byte piece
typedef __attribute__((ext_vector_type(8)))  float  v8f;     // C/D 16x16 f32 accumulator

#define MBLK    128          // rows per workgroup
#define HID     256
#define LDH     264          // bf16 element stride for LDS activation rows (16B-aligned, bank-spread)
#define ALPHA   0.1f

// ---------------- workspace layout (bytes) ----------------
#define WS_C1    0                       // 256 f32
#define WS_W1T   1024                    // 256 x 32  bf16
#define WS_W2T   (WS_W1T + 256*32*2)     // 256 x 256 bf16
#define WS_W3T   (WS_W2T + 256*256*2)
#define WS_W4T   (WS_W3T + 256*256*2)
#define WS_W5T   (WS_W4T + 256*256*2)    // 16 x 256 bf16
#define WS_END   (WS_W5T + 16*256*2)

// ---------------------------------------------------------------------------
// CDNA5 hardware tanh (TRANS op).
// ---------------------------------------------------------------------------
__device__ __forceinline__ float hw_tanhf(float x) {
#if __has_builtin(__builtin_amdgcn_tanhf)
    return __builtin_amdgcn_tanhf(x);
#else
    float y;
    asm("v_tanh_f32 %0, %1\n\tv_nop" : "=v"(y) : "v"(x));
    return y;
#endif
}

__device__ __forceinline__ float gelu_fast(float x) {
    // tanh-form GELU: 0.5*x*(1 + tanh(sqrt(2/pi)*(x + 0.044715*x^3)))
    float x3 = x * x * x;
    float t  = hw_tanhf(fmaf(x3, 0.0356774081f, x * 0.7978845608f));
    return 0.5f * x * (1.0f + t);
}

// ---------------------------------------------------------------------------
// Prep: fold latent into layer-1 bias:  c1[n] = b1[n] + sum_e lat[e]*W1[30+e][n]
// ---------------------------------------------------------------------------
__global__ void prep_c1(const float* __restrict__ traj_w, const int* __restrict__ traj_ids,
                        const float* __restrict__ W1, const float* __restrict__ b1,
                        float* __restrict__ c1) {
    int n = threadIdx.x;                      // 256 threads
    const float* lat = traj_w + (size_t)traj_ids[0] * 64;
    float acc = b1[n];
    #pragma unroll 8
    for (int e = 0; e < 64; ++e)
        acc += lat[e] * W1[(size_t)(30 + e) * 256 + n];
    c1[n] = acc;
}

// ---------------------------------------------------------------------------
// Prep: transpose + bf16-convert weights: Wt[n][k] = (k<Kuse && n<Nsrc) ? W[k][n] : 0
// ---------------------------------------------------------------------------
__global__ void prep_wt(const float* __restrict__ W, bf16_t* __restrict__ Wt,
                        int Kuse, int Nsrc, int NT, int KP) {
    int idx = blockIdx.x * blockDim.x + threadIdx.x;
    if (idx >= NT * KP) return;
    int n = idx / KP;
    int k = idx - n * KP;
    float v = (k < Kuse && n < Nsrc) ? W[(size_t)k * Nsrc + n] : 0.0f;
    Wt[idx] = (bf16_t)v;
}

// ---------------------------------------------------------------------------
// Fragment loaders. Layout per CDNA5 ISA §7.12.2 (16-bit, 16x32):
//   lane L, group g=L>>4: VGPR0-3 hold K = kb+8g .. kb+8g+7,
//                          VGPR4-7 hold K = kb+8g+16 .. kb+8g+23.
// A: row index = tile row + (L&15).  B (K x N): column index = tile col + (L&15).
// ---------------------------------------------------------------------------
__device__ __forceinline__ bfrag load_frag(const bf16_t* p) {
    bhalf8 lo = *(const bhalf8*)p;
    bhalf8 hi = *(const bhalf8*)(p + 16);
    bfrag f;
    #pragma unroll
    for (int i = 0; i < 8; ++i) { f[i] = lo[i]; f[i + 8] = hi[i]; }
    return f;
}

__device__ __forceinline__ const bf16_t* frag_addr(const bf16_t* base, int ld,
                                                   int row0, int lane, int kb) {
    return base + (size_t)(row0 + (lane & 15)) * ld + kb + ((lane >> 4) << 3);
}

// ---------------------------------------------------------------------------
// One hidden layer: Out[MBLK x 256] = gelu(A[MBLK x 32*KSTEPS] @ Wt^T + bias)
// Wave w owns n-tiles {2w, 2w+1}, fused: one A-fragment load feeds two WMMAs.
// ---------------------------------------------------------------------------
template <int KSTEPS>
__device__ __forceinline__ void mlp_layer(const bf16_t* __restrict__ Ain, int lda,
                                          bf16_t* __restrict__ Hout, int ldo,
                                          const bf16_t* __restrict__ Wt,
                                          const float* __restrict__ bias,
                                          int lane, int wave) {
    const int nt0 = wave * 2;
    const int nl  = lane & 15;
    const float bn0 = bias[nt0 * 16 + nl];
    const float bn1 = bias[nt0 * 16 + 16 + nl];
    bfrag Bf0[KSTEPS], Bf1[KSTEPS];
    #pragma unroll
    for (int ks = 0; ks < KSTEPS; ++ks) {
        Bf0[ks] = load_frag(frag_addr(Wt, KSTEPS * 32, nt0 * 16,      lane, ks * 32));
        Bf1[ks] = load_frag(frag_addr(Wt, KSTEPS * 32, nt0 * 16 + 16, lane, ks * 32));
    }
    #pragma unroll
    for (int mt = 0; mt < MBLK / 16; ++mt) {
        v8f acc0 = {}, acc1 = {};
        #pragma unroll
        for (int ks = 0; ks < KSTEPS; ++ks) {
            bfrag a = load_frag(frag_addr(Ain, lda, mt * 16, lane, ks * 32));
            acc0 = __builtin_amdgcn_wmma_f32_16x16x32_bf16(
                       false, a, false, Bf0[ks], (short)0, acc0, false, false);
            acc1 = __builtin_amdgcn_wmma_f32_16x16x32_bf16(
                       false, a, false, Bf1[ks], (short)0, acc1, false, false);
        }
        const int mBase = mt * 16 + ((lane >> 4) << 3);
        const int n0 = nt0 * 16 + nl;
        #pragma unroll
        for (int i = 0; i < 8; ++i) {
            bf16_t* hrow = Hout + (size_t)(mBase + i) * ldo + n0;
            hrow[0]  = (bf16_t)gelu_fast(acc0[i] + bn0);
            hrow[16] = (bf16_t)gelu_fast(acc1[i] + bn1);
        }
    }
}

// Output layer: OutLds[MBLK x 16] (f32) = A[MBLK x 256] @ W5t^T + b5 (N padded 9->16)
__device__ __forceinline__ void mlp_out_layer(const bf16_t* __restrict__ Ain, int lda,
                                              float* __restrict__ OutLds,
                                              const bf16_t* __restrict__ Wt,
                                              const float* __restrict__ b5,
                                              int lane, int wave) {
    const int mt = wave;                       // 8 waves, 8 M-tiles
    const int nl = lane & 15;
    const float bn = (nl < 9) ? b5[nl] : 0.0f;
    bfrag Bf[8];
    #pragma unroll
    for (int ks = 0; ks < 8; ++ks)
        Bf[ks] = load_frag(frag_addr(Wt, 256, 0, lane, ks * 32));
    v8f acc = {};
    #pragma unroll
    for (int ks = 0; ks < 8; ++ks) {
        bfrag a = load_frag(frag_addr(Ain, lda, mt * 16, lane, ks * 32));
        acc = __builtin_amdgcn_wmma_f32_16x16x32_bf16(
                  false, a, false, Bf[ks], (short)0, acc, false, false);
    }
    const int mBase = mt * 16 + ((lane >> 4) << 3);
    #pragma unroll
    for (int i = 0; i < 8; ++i)
        OutLds[(mBase + i) * 16 + nl] = acc[i] + bn;
}

// ---------------------------------------------------------------------------
// Main kernel: SVD features -> 5-layer WMMA MLP -> polar update.
// 256 threads = 8 wave32; 128 rows per block.
// ---------------------------------------------------------------------------
__global__ __launch_bounds__(256, 2)
void fproj_main(const float* __restrict__ F,
                const bf16_t* __restrict__ W1t, const float* __restrict__ c1,
                const bf16_t* __restrict__ W2t, const float* __restrict__ b2,
                const bf16_t* __restrict__ W3t, const float* __restrict__ b3,
                const bf16_t* __restrict__ W4t, const float* __restrict__ b4,
                const bf16_t* __restrict__ W5t, const float* __restrict__ b5,
                float* __restrict__ out) {
    __shared__ __align__(16) bf16_t s_feat[MBLK * 32];
    __shared__ __align__(16) bf16_t s_hA[MBLK * LDH];
    __shared__ __align__(16) bf16_t s_hB[MBLK * LDH];

    const int tid  = threadIdx.x;
    const int lane = tid & 31;
    const int wave = tid >> 5;
    const size_t row0 = (size_t)blockIdx.x * MBLK;

    float Ff[3][3];   // original F, kept in registers for epilogue
    float Rm[3][3];   // polar rotation R = U * Vh

    // ---------- per-row 3x3 SVD via one-sided Jacobi (threads 0..127) ----------
    if (tid < MBLK) {
        const float* fp = F + (row0 + tid) * 9;
        float Aw[3][3], Vv[3][3], Uu[3][3], sig[3];
        #pragma unroll
        for (int i = 0; i < 3; ++i)
            #pragma unroll
            for (int j = 0; j < 3; ++j) {
                Ff[i][j] = fp[i * 3 + j];
                Aw[i][j] = Ff[i][j];
                Vv[i][j] = (i == j) ? 1.0f : 0.0f;
            }

        auto rot = [&](int p, int q) {
            float app = 0.f, aqq = 0.f, apq = 0.f;
            #pragma unroll
            for (int i = 0; i < 3; ++i) {
                float x = Aw[i][p], y = Aw[i][q];
                app += x * x; aqq += y * y; apq += x * y;
            }
            if (fabsf(apq) > 1e-20f) {
                float zeta = (aqq - app) / (2.0f * apq);
                float tt = copysignf(1.0f, zeta) / (fabsf(zeta) + sqrtf(1.0f + zeta * zeta));
                float cs = 1.0f / sqrtf(1.0f + tt * tt);
                float sn = cs * tt;
                #pragma unroll
                for (int i = 0; i < 3; ++i) {
                    float x = Aw[i][p], y = Aw[i][q];
                    Aw[i][p] = cs * x - sn * y;  Aw[i][q] = sn * x + cs * y;
                    float vx = Vv[i][p], vy = Vv[i][q];
                    Vv[i][p] = cs * vx - sn * vy; Vv[i][q] = sn * vx + cs * vy;
                }
            }
        };
        #pragma unroll
        for (int sweep = 0; sweep < 4; ++sweep) { rot(0, 1); rot(0, 2); rot(1, 2); }

        #pragma unroll
        for (int j = 0; j < 3; ++j) {
            float s2 = Aw[0][j]*Aw[0][j] + Aw[1][j]*Aw[1][j] + Aw[2][j]*Aw[2][j];
            sig[j] = sqrtf(s2);
            float inv = 1.0f / fmaxf(sig[j], 1e-20f);
            #pragma unroll
            for (int i = 0; i < 3; ++i) Uu[i][j] = Aw[i][j] * inv;
        }
        auto swapc = [&](int a, int b) {
            float s = sig[a]; sig[a] = sig[b]; sig[b] = s;
            #pragma unroll
            for (int i = 0; i < 3; ++i) {
                float u = Uu[i][a]; Uu[i][a] = Uu[i][b]; Uu[i][b] = u;
                float v = Vv[i][a]; Vv[i][a] = Vv[i][b]; Vv[i][b] = v;
            }
        };
        if (sig[0] < sig[1]) swapc(0, 1);
        if (sig[1] < sig[2]) swapc(1, 2);
        if (sig[0] < sig[1]) swapc(0, 1);

        // R = U * Vh = U * V^T
        #pragma unroll
        for (int i = 0; i < 3; ++i)
            #pragma unroll
            for (int j = 0; j < 3; ++j)
                Rm[i][j] = Uu[i][0]*Vv[j][0] + Uu[i][1]*Vv[j][1] + Uu[i][2]*Vv[j][2];

        // features: [F-I (9) | U-I (9) | sig-1 (3) | Vh-I (9) | pad (2)]
        bf16_t* fr = s_feat + tid * 32;
        #pragma unroll
        for (int i = 0; i < 3; ++i)
            #pragma unroll
            for (int j = 0; j < 3; ++j) {
                float d = (i == j) ? 1.0f : 0.0f;
                fr[i * 3 + j]      = (bf16_t)(Ff[i][j] - d);
                fr[9 + i * 3 + j]  = (bf16_t)(Uu[i][j] - d);
                fr[21 + i * 3 + j] = (bf16_t)(Vv[j][i] - d);   // Vh[i][j] = V[j][i]
            }
        #pragma unroll
        for (int j = 0; j < 3; ++j) fr[18 + j] = (bf16_t)(sig[j] - 1.0f);
        fr[30] = (bf16_t)0.0f;  fr[31] = (bf16_t)0.0f;
    }
    __syncthreads();

    // ---------- 5-layer MLP on WMMA ----------
    mlp_layer<1>(s_feat, 32,  s_hA, LDH, W1t, c1, lane, wave);  __syncthreads();
    mlp_layer<8>(s_hA,  LDH,  s_hB, LDH, W2t, b2, lane, wave);  __syncthreads();
    mlp_layer<8>(s_hB,  LDH,  s_hA, LDH, W3t, b3, lane, wave);  __syncthreads();
    mlp_layer<8>(s_hA,  LDH,  s_hB, LDH, W4t, b4, lane, wave);  __syncthreads();

    float* s_outf = (float*)s_hA;   // layer-4 input is dead; reuse as f32 [MBLK x 16]
    mlp_out_layer(s_hB, LDH, s_outf, W5t, b5, lane, wave);
    __syncthreads();

    // ---------- epilogue: symmetrize, Fp = ALPHA * R @ x + F ----------
    if (tid < MBLK) {
        const float* orow = s_outf + tid * 16;
        float x[9];
        #pragma unroll
        for (int j = 0; j < 9; ++j) x[j] = orow[j];
        float S[3][3];
        #pragma unroll
        for (int i = 0; i < 3; ++i)
            #pragma unroll
            for (int j = 0; j < 3; ++j)
                S[i][j] = 0.5f * (x[i * 3 + j] + x[j * 3 + i]);
        float* op = out + (row0 + tid) * 9;
        #pragma unroll
        for (int i = 0; i < 3; ++i)
            #pragma unroll
            for (int j = 0; j < 3; ++j)
                op[i * 3 + j] = ALPHA * (Rm[i][0]*S[0][j] + Rm[i][1]*S[1][j] + Rm[i][2]*S[2][j])
                                + Ff[i][j];
    }
}

// ---------------------------------------------------------------------------
// Host launcher
// ---------------------------------------------------------------------------
extern "C" void kernel_launch(void* const* d_in, const int* in_sizes, int n_in,
                              void* d_out, int out_size, void* d_ws, size_t ws_size,
                              hipStream_t stream) {
    const float* F      = (const float*)d_in[0];
    const float* traj_w = (const float*)d_in[1];
    const float* W1     = (const float*)d_in[2];
    const float* b1     = (const float*)d_in[3];
    const float* W2     = (const float*)d_in[4];
    const float* b2     = (const float*)d_in[5];
    const float* W3     = (const float*)d_in[6];
    const float* b3     = (const float*)d_in[7];
    const float* W4     = (const float*)d_in[8];
    const float* b4     = (const float*)d_in[9];
    const float* W5     = (const float*)d_in[10];
    const float* b5     = (const float*)d_in[11];
    const int*   tids   = (const int*)d_in[12];
    float* out = (float*)d_out;

    unsigned char* ws = (unsigned char*)d_ws;
    float*  c1  = (float*)(ws + WS_C1);
    bf16_t* W1t = (bf16_t*)(ws + WS_W1T);
    bf16_t* W2t = (bf16_t*)(ws + WS_W2T);
    bf16_t* W3t = (bf16_t*)(ws + WS_W3T);
    bf16_t* W4t = (bf16_t*)(ws + WS_W4T);
    bf16_t* W5t = (bf16_t*)(ws + WS_W5T);

    // weight prep (tiny; all L2-resident)
    prep_c1<<<1, 256, 0, stream>>>(traj_w, tids, W1, b1, c1);
    prep_wt<<<(256 * 32   + 255) / 256, 256, 0, stream>>>(W1, W1t,  30, 256, 256, 32);
    prep_wt<<<(256 * 256  + 255) / 256, 256, 0, stream>>>(W2, W2t, 256, 256, 256, 256);
    prep_wt<<<(256 * 256  + 255) / 256, 256, 0, stream>>>(W3, W3t, 256, 256, 256, 256);
    prep_wt<<<(256 * 256  + 255) / 256, 256, 0, stream>>>(W4, W4t, 256, 256, 256, 256);
    prep_wt<<<(16  * 256  + 255) / 256, 256, 0, stream>>>(W5, W5t, 256,   9,  16, 256);

    const int rows   = in_sizes[0] / 9;        // B = 262144
    const int blocks = rows / MBLK;            // 2048
    fproj_main<<<blocks, 256, 0, stream>>>(F, W1t, c1, W2t, b2, W3t, b3,
                                           W4t, b4, W5t, b5, out);
}
